// GAE1_58866821759633
// MI455X (gfx1250) — matrix operand order repaired and checked by reading, not compiled, not run on hardware
//
#include <hip/hip_runtime.h>

// ---- problem constants (match reference) ----
#define NN   102400   // total nodes
#define BB   1024     // graphs
#define NPGc 100      // nodes per graph
#define EPGc 1000     // edges per graph (edge list is graph-contiguous by construction)
#define EE   1024000  // edges
#define INCc 100
#define HIDc 128
#define INDc 64

typedef __attribute__((ext_vector_type(16))) __bf16 v16bf;
typedef __attribute__((ext_vector_type(8)))  __bf16 v8bf;
typedef __attribute__((ext_vector_type(8)))  float  v8f;

static inline int ceildiv(int a, int b) { return (a + b - 1) / b; }

__device__ __forceinline__ unsigned short f2bf(float f) {
    unsigned u = __float_as_uint(f);
    u += 0x7FFFu + ((u >> 16) & 1u);          // round-to-nearest-even
    return (unsigned short)(u >> 16);
}

// ---------------- per-graph GCN normalization (LDS, no global atomics) -----
__global__ void k_norm_graph(const int* ei, float* selfw, float* ne) {
    __shared__ float s_deg[NPGc];
    __shared__ float s_dinv[NPGc];
    __shared__ int   s_s[EPGc];
    __shared__ int   s_d[EPGc];
    int b = blockIdx.x, tid = threadIdx.x, base = b * NPGc;
    for (int i = tid; i < NPGc; i += blockDim.x) s_deg[i] = 0.0f;
    for (int e = tid; e < EPGc; e += blockDim.x) {
        int eg = b * EPGc + e;
        s_s[e] = ei[eg] - base;
        s_d[e] = ei[EE + eg] - base;
    }
    __syncthreads();
    for (int e = tid; e < EPGc; e += blockDim.x) atomicAdd(&s_deg[s_d[e]], 1.0f);
    __syncthreads();
    for (int i = tid; i < NPGc; i += blockDim.x) {
        float r = rsqrtf(s_deg[i] + 1.0f);     // +1 self loop
        s_dinv[i] = r;
        selfw[base + i] = r * r;
    }
    __syncthreads();
    for (int e = tid; e < EPGc; e += blockDim.x)
        ne[b * EPGc + e] = s_dinv[s_s[e]] * s_dinv[s_d[e]];
}

// ---- fused GCN aggregate+finalize: one workgroup per graph, LDS atomics ----
// out[n,f] = act(tanh(sum_edges norm*lin[src,f] + lin[n,f]*selfw[n] + bias[f]))
template <int F>
__global__ void k_gcn_graph(const int* ei, const float* ne, const float* lin,
                            const float* selfw, const float* bias, float* out, int act) {
    extern __shared__ float smem[];
    float* s_lin = smem;                   // 100*F
    float* s_agg = smem + NPGc * F;        // 100*F
    float* s_ne  = smem + 2 * NPGc * F;    // 1000
    int*   s_src = (int*)(s_ne + EPGc);    // 1000
    int*   s_dst = s_src + EPGc;           // 1000
    int b = blockIdx.x, tid = threadIdx.x, base = b * NPGc;

    for (int e = tid; e < EPGc; e += blockDim.x) {
        int eg = b * EPGc + e;
        s_src[e] = (ei[eg] - base) * F;
        s_dst[e] = (ei[EE + eg] - base) * F;
        s_ne[e]  = ne[eg];
    }
    for (int i = tid; i < NPGc * F; i += blockDim.x) {
        s_lin[i] = lin[(size_t)base * F + i];
        s_agg[i] = 0.0f;
    }
    __syncthreads();
    for (int idx = tid; idx < EPGc * F; idx += blockDim.x) {
        int e = idx / F, f = idx - e * F;
        atomicAdd(&s_agg[s_dst[e] + f], s_lin[s_src[e] + f] * s_ne[e]);
    }
    __syncthreads();
    for (int i = tid; i < NPGc * F; i += blockDim.x) {
        int node = i / F, f = i - node * F;
        float v = tanhf(s_agg[i] + s_lin[i] * selfw[base + node] + bias[f]);
        if (act == 1) v = (v > 0.0f) ? v : 0.01f * v;
        out[(size_t)base * F + i] = v;
    }
}

// ---------------- precision/layout conversion kernels ----------------
// f32 [rows x cols] -> bf16 [rows x Kpad], zero padded
__global__ void k_cvt_pad(const float* src, unsigned short* dst, int rows, int cols, int Kpad) {
    int i = blockIdx.x * blockDim.x + threadIdx.x;
    int total = rows * Kpad;
    if (i >= total) return;
    int r = i / Kpad, k = i - r * Kpad;
    dst[i] = (k < cols) ? f2bf(src[(size_t)r * cols + k]) : (unsigned short)0;
}

// W [K x F] f32 -> Wt [Fpad x Kpad] bf16 (B-fragment friendly), zero padded
__global__ void k_wt(const float* W, unsigned short* Wt, int K, int F, int Kpad, int Fpad) {
    int i = blockIdx.x * blockDim.x + threadIdx.x;
    int total = Fpad * Kpad;
    if (i >= total) return;
    int f = i / Kpad, k = i - f * Kpad;
    Wt[i] = (f < F && k < K) ? f2bf(W[(size_t)k * F + f]) : (unsigned short)0;
}

// ---------------- WMMA fragment loaders (ISA 16-bit layouts) ----------------
__device__ __forceinline__ v16bf cat8(v8bf lo, v8bf hi) {
    return __builtin_shufflevector(lo, hi, 0,1,2,3,4,5,6,7,8,9,10,11,12,13,14,15);
}

// A 16x32 bf16, row-major source, lda elements
__device__ __forceinline__ v16bf load_fragA(const __bf16* A, int lda, int row0, int k0, int lane) {
    int m = lane & 15, half = lane >> 4;
    const __bf16* p = A + (size_t)(row0 + m) * lda + k0 + half * 8;
    v8bf lo = *(const v8bf*)p;          // elems 0..7  : K = half*8 + e
    v8bf hi = *(const v8bf*)(p + 16);   // elems 8..15 : K = 16 + half*8 + e
    return cat8(lo, hi);
}

// B 32x16 bf16, supplied transposed: Bt[n][k], ldb elements
__device__ __forceinline__ v16bf load_fragB(const __bf16* Bt, int ldb, int col0, int k0, int lane) {
    int n = lane & 15, half = lane >> 4;
    const __bf16* p = Bt + (size_t)(col0 + n) * ldb + k0 + half * 8;
    v8bf lo = *(const v8bf*)p;          // elems 0..7  : K = half*8 + e
    v8bf hi = *(const v8bf*)(p + 16);   // elems 8..15 : K = 16 + half*8 + e
    return cat8(lo, hi);
}

// C = A(bf16 MxKpad) x Wt(bf16 FpadxKpad)^T-logical -> f32 [M x Fout], col guard
__global__ __launch_bounds__(128)
void k_wmma_gemm(const unsigned short* Abf, const unsigned short* Btbf,
                 float* C, int M, int Kpad, int Fout, int ntiles) {
    int wave = (blockIdx.x * blockDim.x + threadIdx.x) >> 5;
    int lane = threadIdx.x & 31;
    int mt = wave / ntiles, nt = wave - mt * ntiles;
    if (mt * 16 >= M) return;
    const __bf16* A  = (const __bf16*)Abf;
    const __bf16* Bt = (const __bf16*)Btbf;
    int row0 = mt * 16, col0 = nt * 16;
    v8f c = {};
    for (int k0 = 0; k0 < Kpad; k0 += 32) {
        v16bf a = load_fragA(A, Kpad, row0, k0, lane);
        v16bf b = load_fragB(Bt, Kpad, col0, k0, lane);
        c = __builtin_amdgcn_wmma_f32_16x16x32_bf16(false, a, false, b, (short)0, c, false, false);
    }
    int n = lane & 15, half = lane >> 4;
    int col = col0 + n;
    if (col < Fout) {
#pragma unroll
        for (int v = 0; v < 8; ++v) {
            int row = row0 + v + 8 * half;
            C[(size_t)row * Fout + col] = c[v];
        }
    }
}

// readout linear: lin[n] = sum_k h[n,0..63] * Wr[k]
__global__ void k_dot64(const float* h, const float* Wr, float* lin) {
    int n = blockIdx.x * blockDim.x + threadIdx.x;
    if (n >= NN) return;
    const float* hr = h + (size_t)n * 64;
    float s = 0.0f;
#pragma unroll 8
    for (int k = 0; k < 64; ++k) s += hr[k] * Wr[k];
    lin[n] = s;
}

// heads on h [B,100]: sigmoid dis head, softmax y head
__global__ void k_heads(const float* h, const float* Wdis, const float* bdis,
                        const float* Wy, const float* by, float* hdis, float* ypred) {
    int b = blockIdx.x * blockDim.x + threadIdx.x;
    if (b >= BB) return;
    const float* hr = h + (size_t)b * NPGc;
    float s0 = bdis[0], s1 = bdis[1], t0 = by[0], t1 = by[1];
    for (int k = 0; k < NPGc; ++k) {
        float v = hr[k];
        s0 += v * Wdis[2 * k];  s1 += v * Wdis[2 * k + 1];
        t0 += v * Wy[2 * k];    t1 += v * Wy[2 * k + 1];
    }
    hdis[2 * b]     = 1.0f / (1.0f + expf(-s0));
    hdis[2 * b + 1] = 1.0f / (1.0f + expf(-s1));
    float m = fmaxf(t0, t1);
    float e0 = expf(t0 - m), e1 = expf(t1 - m), si = e0 + e1;
    ypred[2 * b]     = e0 / si;
    ypred[2 * b + 1] = e1 / si;
}

// mu/logvar bias-add, reparameterize
__global__ void k_z1(const float* gm, const float* gv, const float* bmu, const float* bvar,
                     const float* eps, float* muo, float* lvo, float* z1) {
    int i = blockIdx.x * blockDim.x + threadIdx.x;
    if (i >= BB * INDc) return;
    int f = i & 63;
    float mu = gm[i] + bmu[f];
    float lv = gv[i] + bvar[f];
    muo[i] = mu;
    lvo[i] = lv;
    z1[i] = eps[i] * expf(0.5f * lv) + mu;
}

__global__ void k_addb100(float* z, const float* bzd) {
    int i = blockIdx.x * blockDim.x + threadIdx.x;
    if (i < NN) z[i] += bzd[i % NPGc];
}

// decoder first layer linear: lin[n,f] = z[n] * Wrr[f]  (K=1)
__global__ void k_outer(const float* z, const float* Wrr, float* lin) {
    int i = blockIdx.x * blockDim.x + threadIdx.x;
    if (i >= NN * 64) return;
    int n = i >> 6, f = i & 63;
    lin[i] = z[n] * Wrr[f];
}

// xd [N,100] f32 -> padded per-graph bf16 [B][112][128] (zeros outside 100x100)
__global__ void k_padg(const float* xd, unsigned short* xpad) {
    int i = blockIdx.x * blockDim.x + threadIdx.x;
    const int total = BB * 112 * 128;
    if (i >= total) return;
    int b = i / (112 * 128);
    int r = i - b * (112 * 128);
    int row = r >> 7, j = r & 127;
    xpad[i] = (row < NPGc && j < NPGc)
                  ? f2bf(xd[((size_t)b * NPGc + row) * NPGc + j])
                  : (unsigned short)0;
}

// batched Gram: out[b,i,k] = sum_j X[b,i,j]*X[b,k,j]; zero diagonal
__global__ __launch_bounds__(128)
void k_gram(const unsigned short* xpad, float* out) {
    int wave = (blockIdx.x * blockDim.x + threadIdx.x) >> 5;
    int lane = threadIdx.x & 31;
    const int TT = 7 * 7;
    int b = wave / TT;
    if (b >= BB) return;
    int t = wave - b * TT;
    int it = t / 7, kt = t - it * 7;
    const __bf16* X = (const __bf16*)xpad + (size_t)b * 112 * 128;
    int row0 = it * 16, col0 = kt * 16;
    v8f c = {};
    for (int k0 = 0; k0 < 128; k0 += 32) {
        v16bf a  = load_fragA(X, 128, row0, k0, lane);
        v16bf bf = load_fragB(X, 128, col0, k0, lane);
        c = __builtin_amdgcn_wmma_f32_16x16x32_bf16(false, a, false, bf, (short)0, c, false, false);
    }
    int n = lane & 15, half = lane >> 4;
    int col = col0 + n;
    if (col < NPGc) {
#pragma unroll
        for (int v = 0; v < 8; ++v) {
            int i = row0 + v + 8 * half;
            if (i < NPGc) {
                out[((size_t)b * NPGc + i) * NPGc + col] = (i == col) ? 0.0f : c[v];
            }
        }
    }
}

// ---------------- host orchestration ----------------
extern "C" void kernel_launch(void* const* d_in, const int* in_sizes, int n_in,
                              void* d_out, int out_size, void* d_ws, size_t ws_size,
                              hipStream_t stream) {
    const float* x    = (const float*)d_in[0];
    const int*   ei   = (const int*)d_in[1];
    // d_in[2] edge_attr, d_in[3] pcd, d_in[4] eyes: unused by reference math
    const float* eps  = (const float*)d_in[5];
    const float* W1   = (const float*)d_in[6];
    const float* b1   = (const float*)d_in[7];
    const float* W2   = (const float*)d_in[8];
    const float* b2   = (const float*)d_in[9];
    const float* Wr   = (const float*)d_in[10];
    const float* br   = (const float*)d_in[11];
    const float* Wdis = (const float*)d_in[12];
    const float* bdis = (const float*)d_in[13];
    const float* Wy   = (const float*)d_in[14];
    const float* by   = (const float*)d_in[15];
    const float* Wmu  = (const float*)d_in[16];
    const float* bmu  = (const float*)d_in[17];
    const float* Wvar = (const float*)d_in[18];
    const float* bvar = (const float*)d_in[19];
    const float* Wzd  = (const float*)d_in[20];
    const float* bzd  = (const float*)d_in[21];
    const float* Wrr  = (const float*)d_in[22];
    const float* brr  = (const float*)d_in[23];
    const float* Wu1  = (const float*)d_in[24];
    const float* bu1  = (const float*)d_in[25];
    const float* Wu2  = (const float*)d_in[26];
    const float* bu2  = (const float*)d_in[27];

    float* o_out  = (float*)d_out;
    float* o_hdis = o_out + (size_t)NN * NPGc;
    float* o_y    = o_hdis + BB * 2;
    float* o_mu   = o_y + BB * 2;
    float* o_lv   = o_mu + BB * INDc;

    // workspace carve-out (256B aligned regions)
    size_t off = 0;
    auto carve = [&](size_t bytes) -> char* {
        char* p = (char*)d_ws + off;
        off += (bytes + 255) & ~(size_t)255;
        return p;
    };
    float* selfw = (float*)carve((size_t)NN * 4);
    float* norme = (float*)carve((size_t)EE * 4);
    float* bufA  = (float*)carve((size_t)NN * 128 * 4);
    unsigned short* bfbuf = (unsigned short*)carve((size_t)BB * 112 * 128 * 2); // also xpad
    float* vecA  = (float*)carve((size_t)NN * 4);
    float* vecB  = (float*)carve((size_t)NN * 4);
    float* gm    = (float*)carve((size_t)BB * 64 * 4);
    float* gv    = (float*)carve((size_t)BB * 64 * 4);
    float* z1f   = (float*)carve((size_t)BB * 64 * 4);
    unsigned short* W1t  = (unsigned short*)carve(128 * 128 * 2);
    unsigned short* W2t  = (unsigned short*)carve(64 * 128 * 2);
    unsigned short* Wmut = (unsigned short*)carve(64 * 128 * 2);
    unsigned short* Wvt  = (unsigned short*)carve(64 * 128 * 2);
    unsigned short* Wzdt = (unsigned short*)carve(112 * 64 * 2);
    unsigned short* Wu1t = (unsigned short*)carve(128 * 64 * 2);
    unsigned short* Wu2t = (unsigned short*)carve(112 * 128 * 2);
    (void)ws_size; (void)in_sizes; (void)n_in; (void)out_size;

    const int T = 256;
#define GRID(n) dim3(ceildiv((n), T)), dim3(T), 0, stream
    auto gemm = [&](const unsigned short* A, const unsigned short* Bt, float* C,
                    int M, int Kpad, int Fout, int ntiles) {
        int waves = (M / 16) * ntiles;
        int blocks = ceildiv(waves * 32, 128);
        k_wmma_gemm<<<dim3(blocks), dim3(128), 0, stream>>>(A, Bt, C, M, Kpad, Fout, ntiles);
    };
    // dynamic LDS bytes for fused GCN kernel: 2*100*F lin/agg + 1000 ne + 2000 src/dst
    auto gcn_shmem = [](int F) -> size_t { return (size_t)(800 * F + 12000); };

    // per-graph symmetric normalization (LDS accumulation, no global atomics)
    k_norm_graph<<<dim3(BB), dim3(T), 0, stream>>>(ei, selfw, norme);

    // weight prep (transpose + bf16 + pad)
    k_wt<<<GRID(128 * 128)>>>(W1, W1t, 100, 128, 128, 128);
    k_wt<<<GRID(64 * 128)>>>(W2, W2t, 128, 64, 128, 64);
    k_wt<<<GRID(64 * 128)>>>(Wmu, Wmut, 100, 64, 128, 64);
    k_wt<<<GRID(64 * 128)>>>(Wvar, Wvt, 100, 64, 128, 64);
    k_wt<<<GRID(112 * 64)>>>(Wzd, Wzdt, 64, 100, 64, 112);
    k_wt<<<GRID(128 * 64)>>>(Wu1, Wu1t, 64, 128, 64, 128);
    k_wt<<<GRID(112 * 128)>>>(Wu2, Wu2t, 128, 100, 128, 112);

    // ---- Encoder L1: GCN 100 -> 128, leaky(tanh) ----
    k_cvt_pad<<<GRID(NN * 128)>>>(x, bfbuf, NN, 100, 128);
    gemm(bfbuf, W1t, bufA, NN, 128, 128, 8);
    k_gcn_graph<128><<<dim3(BB), dim3(T), gcn_shmem(128), stream>>>(ei, norme, bufA, selfw, b1, bufA, 1);

    // ---- Encoder L2: GCN 128 -> 64, leaky(tanh) ----
    k_cvt_pad<<<GRID(NN * 128)>>>(bufA, bfbuf, NN, 128, 128);
    gemm(bfbuf, W2t, bufA, NN, 128, 64, 4);
    k_gcn_graph<64><<<dim3(BB), dim3(T), gcn_shmem(64), stream>>>(ei, norme, bufA, selfw, b2, bufA, 1);

    // ---- Readout conv: GCN 64 -> 1, tanh ----
    k_dot64<<<GRID(NN)>>>(bufA, Wr, vecA);
    k_gcn_graph<1><<<dim3(BB), dim3(T), gcn_shmem(1), stream>>>(ei, norme, vecA, selfw, br, vecA, 0);
    // h = vecA [B,100]

    // ---- heads ----
    k_heads<<<GRID(BB)>>>(vecA, Wdis, bdis, Wy, by, o_hdis, o_y);
    k_cvt_pad<<<GRID(BB * 128)>>>(vecA, bfbuf, BB, 100, 128);
    gemm(bfbuf, Wmut, gm, BB, 128, 64, 4);
    gemm(bfbuf, Wvt, gv, BB, 128, 64, 4);
    k_z1<<<GRID(BB * 64)>>>(gm, gv, bmu, bvar, eps, o_mu, o_lv, z1f);
    k_cvt_pad<<<GRID(BB * 64)>>>(z1f, bfbuf, BB, 64, 64);
    gemm(bfbuf, Wzdt, vecB, BB, 64, 100, 7);
    k_addb100<<<GRID(NN)>>>(vecB, bzd);                 // z = vecB [N]

    // ---- Decoder L0: GCN 1 -> 64, tanh ----
    k_outer<<<GRID(NN * 64)>>>(vecB, Wrr, bufA);
    k_gcn_graph<64><<<dim3(BB), dim3(T), gcn_shmem(64), stream>>>(ei, norme, bufA, selfw, brr, bufA, 0);

    // ---- Decoder L1: GCN 64 -> 128, tanh ----
    k_cvt_pad<<<GRID(NN * 64)>>>(bufA, bfbuf, NN, 64, 64);
    gemm(bfbuf, Wu1t, bufA, NN, 64, 128, 8);
    k_gcn_graph<128><<<dim3(BB), dim3(T), gcn_shmem(128), stream>>>(ei, norme, bufA, selfw, bu1, bufA, 0);

    // ---- Decoder L2: GCN 128 -> 100, tanh ----
    k_cvt_pad<<<GRID(NN * 128)>>>(bufA, bfbuf, NN, 128, 128);
    gemm(bfbuf, Wu2t, bufA, NN, 128, 100, 7);
    k_gcn_graph<100><<<dim3(BB), dim3(T), gcn_shmem(100), stream>>>(ei, norme, bufA, selfw, bu2, bufA, 0);

    // ---- batched Gram X X^T with zero diagonal ----
    k_padg<<<GRID(BB * 112 * 128)>>>(bufA, bfbuf);
    {
        int waves = BB * 7 * 7;
        int blocks = ceildiv(waves * 32, 128);
        k_gram<<<dim3(blocks), dim3(128), 0, stream>>>(bfbuf, o_out);
    }
#undef GRID
}
